// GridTorch_27556510171346
// MI455X (gfx1250) — matrix operand order repaired and checked by reading, not compiled, not run on hardware
//
#include <hip/hip_runtime.h>
#include <hip/hip_bf16.h>

typedef __attribute__((ext_vector_type(16))) _Float16 v16h;
typedef __attribute__((ext_vector_type(8)))  float    v8f;

// global (addrspace 1) flavors so weight loads lower to global_load_b128,
// not flat_load_b128 (FLAT would tie up DScnt alongside LOADcnt).
typedef _Float16 __attribute__((address_space(1))) f16g;
typedef v16h     __attribute__((address_space(1))) v16h_g;

constexpr int TSTEPS = 100;
constexpr int BATCH  = 4096;
constexpr int NH     = 128;
constexpr int NPCd   = 256;
constexpr int NHDd   = 12;
constexpr int NBd    = 256;
constexpr int BT     = 64;     // batch rows per block
constexpr int NTHR   = 512;    // 16 wave32s -> 4 waves per SIMD32
constexpr int KA     = 160;    // augmented gate K: 128 h + 3 x + 1 bias + 28 zero-pad

__device__ __forceinline__ float sigmoid_f(float x){ return 1.f/(1.f+__expf(-x)); }
__device__ __forceinline__ float tanh_f(float x){
  // 1 - 2/(e^{2x}+1): saturates to +/-1 as exp under/overflows, no clamp needed
  float e = __expf(2.f*x);
  return 1.f - 2.f/(e+1.f);
}

// Launder a (uniform) pointer through an empty volatile asm so loads based on
// it cannot be hoisted out of the timestep loop (prevents LICM -> scratch
// spill), and return it as an explicit GLOBAL address-space pointer.
__device__ __forceinline__ const f16g* keep(const _Float16* p){
  unsigned long long u = (unsigned long long)p;
  asm volatile("" : "+s"(u));
  return (const f16g*)u;
}

__device__ __forceinline__ v8f wmma_f16(v16h a, v16h b, v8f c){
  // D = A(16x32,f16) * B(32x16,f16) + C(16x16,f32)
  return __builtin_amdgcn_wmma_f32_16x16x32_f16(false, a, false, b, (short)0, c, false, false);
}

// A fragment (16x32 f16, M=lane%16) from a row-major LDS tile [rows x ldk].
// VGPR p holds K = kbase + 2p + 16*(lane>=16) .. +1
// (dwords p=0..7 are one contiguous 32B run)
__device__ __forceinline__ v16h ldsA(const _Float16* tile, int ldk, int m16, int kbase, int lane){
  const int row = m16 + (lane & 15);
  const int hi8 = (lane >> 4) * 16;
  union { v16h v; unsigned u[8]; } un;
  const _Float16* rp = tile + row*ldk + kbase + hi8;
  #pragma unroll
  for (int p = 0; p < 8; ++p){
    const int k = 2*p;
    un.u[p] = *(const unsigned*)(rp + k);
  }
  return un.v;
}

// B fragment (32x16) of B = W^T, where W is row-major f16 [N][K] in global mem.
// lane: N = n16 + lane%16, K = kbase + 16*(lane/16) + j (16 contiguous halves).
__device__ __forceinline__ v16h gB(const f16g* W, int ldk, int n16, int kbase, int lane){
  const int n = n16 + (lane & 15);
  const int k = kbase + ((lane >> 4) << 4);
  return *(const v16h_g*)(W + (size_t)n*ldk + k);
}

// ---------------- prep kernels ----------------
__global__ void cvt_f16(const float* __restrict__ s, _Float16* __restrict__ d, int n){
  int i = blockIdx.x*blockDim.x + threadIdx.x;
  if (i < n) d[i] = (_Float16)s[i];
}
__global__ void cvt_hd_pad(const float* __restrict__ s, _Float16* __restrict__ d){
  int i = blockIdx.x*blockDim.x + threadIdx.x;   // 16*256
  if (i < 16*NBd){
    int r = i >> 8;
    d[i] = (r < NHDd) ? (_Float16)s[i] : (_Float16)0.f;
  }
}
// Augmented gate weight [512 x KA] = [W_hh | W_ih | (b_ih+b_hh) | 0] (f16).
// Column 131 carries the combined bias; the LDS h-tile holds 1.0 there.
__global__ void prep_gate(const float* __restrict__ Whh, const float* __restrict__ Wih,
                          const float* __restrict__ bih, const float* __restrict__ bhh,
                          _Float16* __restrict__ Wga){
  int i = blockIdx.x*blockDim.x + threadIdx.x;   // 512*KA
  if (i < 4*NH*KA){
    int n = i / KA, k = i - n*KA;
    float v = 0.f;
    if (k < NH)            v = Whh[n*NH + k];
    else if (k < NH + 3)   v = Wih[n*3 + (k - NH)];
    else if (k == NH + 3)  v = bih[n] + bhh[n];
    Wga[i] = (_Float16)v;
  }
}
// h0/c0 = [init_pc | init_hd] @ W^T + b   (268-long dots, tiny)
__global__ void init_hc(const float* __restrict__ ipc, const float* __restrict__ ihd,
                        const float* __restrict__ Wst, const float* __restrict__ bst,
                        const float* __restrict__ Wcl, const float* __restrict__ bcl,
                        float* __restrict__ h0, float* __restrict__ c0){
  int gid = blockIdx.x*blockDim.x + threadIdx.x;   // BATCH*NH
  int b = gid >> 7, n = gid & (NH-1);
  float ah = bst[n], ac = bcl[n];
  const float* ip = ipc + (size_t)b*NPCd;
  const float* ih = ihd + (size_t)b*NHDd;
  const float* ws = Wst + (size_t)n*(NPCd+NHDd);
  const float* wc = Wcl + (size_t)n*(NPCd+NHDd);
  for (int d = 0; d < NPCd; ++d){ ah = fmaf(ip[d], ws[d], ah); ac = fmaf(ip[d], wc[d], ac); }
  for (int d = 0; d < NHDd; ++d){ ah = fmaf(ih[d], ws[NPCd+d], ah); ac = fmaf(ih[d], wc[NPCd+d], ac); }
  h0[gid] = ah; c0[gid] = ac;
}

// ---------------- main recurrence ----------------
__global__ __launch_bounds__(NTHR)
void grid_lstm_main(const float* __restrict__ x,
                    const float* __restrict__ b_pc, const float* __restrict__ b_hd,
                    const _Float16* __restrict__ Wg,  const _Float16* __restrict__ Wbn,
                    const _Float16* __restrict__ Wpc, const _Float16* __restrict__ Whd,
                    const float* __restrict__ h0, const float* __restrict__ c0,
                    float* __restrict__ o_hd, float* __restrict__ o_pc,
                    float* __restrict__ o_bn, float* __restrict__ o_h,
                    float* __restrict__ o_c)
{
  __shared__ _Float16 sh_h [BT*KA];   // 20 KB: 0..127 = h, 128..130 = x_t, 131 = 1.0, rest 0
  __shared__ _Float16 sh_bn[BT*NBd];  // 32 KB

  const int tid  = threadIdx.x;
  const int wave = tid >> 5;
  const int lane = tid & 31;
  const int ln   = lane & 15;
  const int hi   = lane >> 4;
  const int gb0  = blockIdx.x * BT;

  const int m  = wave >> 2;    // M-tile (batch-row tile) owned by this wave
  const int q4 = wave & 3;     // quarter of the N space this wave covers

  // stage h0 (f32 -> f16); set bias column to 1.0, zero the other pad columns
  for (int idx = tid; idx < BT*NH; idx += NTHR){
    int row = idx >> 7, col = idx & (NH-1);
    sh_h[row*KA + col] = (_Float16)h0[(size_t)(gb0+row)*NH + col];
  }
  for (int idx = tid; idx < BT*(KA-NH); idx += NTHR){
    int row = idx >> 5, c = idx & 31;
    sh_h[row*KA + NH + c] = (c == 3) ? (_Float16)1.f : (_Float16)0.f;
  }
  // c fragments: wave owns channel tiles ch = 2*q4 + p, p=0..1, at M-tile m
  float creg[2][8];
  #pragma unroll
  for (int p = 0; p < 2; ++p){
    const int ch = 2*q4 + p;
    #pragma unroll
    for (int r = 0; r < 8; ++r){
      const int row = gb0 + m*16 + r + 8*hi;
      creg[p][r] = c0[(size_t)row*NH + ch*16 + ln];
    }
  }
  __syncthreads();

  for (int t = 0; t < TSTEPS; ++t){
    // re-launder weight pointers every step: keeps the B-fragment loads inside
    // the loop (they re-hit L2 for free) instead of hoist+spill to scratch.
    const f16g* WgL  = keep(Wg);
    const f16g* WbnL = keep(Wbn);
    const f16g* WpcL = keep(Wpc);
    const f16g* WhdL = keep(Whd);

    // ---- stage x_t into the augmented K columns (128..130) ----
    if (tid < BT*3){
      int row = tid / 3, d = tid - row*3;
      sh_h[row*KA + NH + d] = (_Float16)x[((size_t)t*BATCH + gb0)*3 + tid];
    }
    __syncthreads();

    // ---- gates = [h|x|1] @ [Whh|Wih|bg]^T : [BT x 512], K = 160 ----
    float hn[2][8];
    #pragma unroll
    for (int p = 0; p < 2; ++p){
      const int ch = 2*q4 + p;
      v8f acc[4];
      #pragma unroll
      for (int q = 0; q < 4; ++q)
        #pragma unroll
        for (int r = 0; r < 8; ++r) acc[q][r] = 0.f;
      // double-buffered B fragments: load kb+1 while multiplying kb
      v16h bf[2][4];
      #pragma unroll
      for (int q = 0; q < 4; ++q) bf[0][q] = gB(WgL, KA, q*NH + ch*16, 0, lane);
      #pragma unroll
      for (int kb = 0; kb < 5; ++kb){
        if (kb < 4){
          #pragma unroll
          for (int q = 0; q < 4; ++q)
            bf[(kb+1)&1][q] = gB(WgL, KA, q*NH + ch*16, (kb+1)*32, lane);
        }
        const v16h a = ldsA(sh_h, KA, m*16, kb*32, lane);
        #pragma unroll
        for (int q = 0; q < 4; ++q)
          acc[q] = wmma_f16(a, bf[kb&1][q], acc[q]);
      }
      #pragma unroll
      for (int r = 0; r < 8; ++r){
        const float ig = sigmoid_f(acc[0][r]);
        const float fg = sigmoid_f(acc[1][r]);
        const float gg = tanh_f   (acc[2][r]);
        const float og = sigmoid_f(acc[3][r]);
        const float cc = fg*creg[p][r] + ig*gg;
        creg[p][r] = cc;
        const float hh = og * tanh_f(cc);
        hn[p][r] = hh;
        const size_t row = (size_t)t*BATCH + gb0 + m*16 + r + 8*hi;
        __builtin_nontemporal_store(hh, &o_h[row*NH + ch*16 + ln]);
        __builtin_nontemporal_store(cc, &o_c[row*NH + ch*16 + ln]);
      }
    }
    __syncthreads();                    // all reads of old sh_h done
    #pragma unroll
    for (int p = 0; p < 2; ++p){
      const int ch = 2*q4 + p;
      #pragma unroll
      for (int r = 0; r < 8; ++r)
        sh_h[(m*16 + r + 8*hi)*KA + ch*16 + ln] = (_Float16)hn[p][r];
    }
    __syncthreads();                    // new sh_h visible

    // ---- bn = tanh(h @ Wbn^T) : [BT x 256], wave covers nt = 4*q4 + i ----
    {
      v8f acc[4];
      #pragma unroll
      for (int i = 0; i < 4; ++i)
        #pragma unroll
        for (int r = 0; r < 8; ++r) acc[i][r] = 0.f;
      v16h bf[2][4];
      #pragma unroll
      for (int i = 0; i < 4; ++i) bf[0][i] = gB(WbnL, NH, (4*q4+i)*16, 0, lane);
      #pragma unroll
      for (int kb = 0; kb < 4; ++kb){
        if (kb < 3){
          #pragma unroll
          for (int i = 0; i < 4; ++i)
            bf[(kb+1)&1][i] = gB(WbnL, NH, (4*q4+i)*16, (kb+1)*32, lane);
        }
        const v16h a = ldsA(sh_h, KA, m*16, kb*32, lane);
        #pragma unroll
        for (int i = 0; i < 4; ++i)
          acc[i] = wmma_f16(a, bf[kb&1][i], acc[i]);
      }
      #pragma unroll
      for (int i = 0; i < 4; ++i){
        const int nt = 4*q4 + i;
        #pragma unroll
        for (int r = 0; r < 8; ++r){
          const float v = tanh_f(acc[i][r]);
          const int lrow = m*16 + r + 8*hi;
          sh_bn[lrow*NBd + nt*16 + ln] = (_Float16)v;
          const size_t row = (size_t)t*BATCH + gb0 + lrow;
          __builtin_nontemporal_store(v, &o_bn[row*NBd + nt*16 + ln]);
        }
      }
    }
    __syncthreads();                    // sh_bn visible

    // ---- pc = bn @ Wpc^T + b_pc : [BT x 256], wave covers nt = 4*q4 + i ----
    {
      v8f acc[4];
      #pragma unroll
      for (int i = 0; i < 4; ++i){
        const float bb = b_pc[(4*q4+i)*16 + ln];
        #pragma unroll
        for (int r = 0; r < 8; ++r) acc[i][r] = bb;
      }
      v16h bf[2][4];
      #pragma unroll
      for (int i = 0; i < 4; ++i) bf[0][i] = gB(WpcL, NBd, (4*q4+i)*16, 0, lane);
      #pragma unroll
      for (int kb = 0; kb < 8; ++kb){
        if (kb < 7){
          #pragma unroll
          for (int i = 0; i < 4; ++i)
            bf[(kb+1)&1][i] = gB(WpcL, NBd, (4*q4+i)*16, (kb+1)*32, lane);
        }
        const v16h a = ldsA(sh_bn, NBd, m*16, kb*32, lane);
        #pragma unroll
        for (int i = 0; i < 4; ++i)
          acc[i] = wmma_f16(a, bf[kb&1][i], acc[i]);
      }
      #pragma unroll
      for (int i = 0; i < 4; ++i){
        #pragma unroll
        for (int r = 0; r < 8; ++r){
          const size_t row = (size_t)t*BATCH + gb0 + m*16 + r + 8*hi;
          __builtin_nontemporal_store(acc[i][r], &o_pc[row*NPCd + (4*q4+i)*16 + ln]);
        }
      }
    }

    // ---- hd = bn @ Whd^T + b_hd (N padded 12->16); waves 12..15, one M-tile each ----
    if (wave >= 12){
      const int mh = wave - 12;
      v8f acc;
      const float bb = (ln < NHDd) ? b_hd[ln] : 0.f;
      #pragma unroll
      for (int r = 0; r < 8; ++r) acc[r] = bb;
      v16h bf[2];
      bf[0] = gB(WhdL, NBd, 0, 0, lane);
      #pragma unroll
      for (int kb = 0; kb < 8; ++kb){
        if (kb < 7) bf[(kb+1)&1] = gB(WhdL, NBd, 0, (kb+1)*32, lane);
        const v16h a = ldsA(sh_bn, NBd, mh*16, kb*32, lane);
        acc = wmma_f16(a, bf[kb&1], acc);
      }
      if (ln < NHDd){
        #pragma unroll
        for (int r = 0; r < 8; ++r){
          const size_t row = (size_t)t*BATCH + gb0 + mh*16 + r + 8*hi;
          __builtin_nontemporal_store(acc[r], &o_hd[row*NHDd + ln]);
        }
      }
    }
    // next-iteration top barrier protects sh_h / sh_bn reuse
  }
}

extern "C" void kernel_launch(void* const* d_in, const int* in_sizes, int n_in,
                              void* d_out, int out_size, void* d_ws, size_t ws_size,
                              hipStream_t stream)
{
  (void)in_sizes; (void)n_in; (void)out_size; (void)ws_size;
  const float* x       = (const float*)d_in[0];
  const float* init_pc = (const float*)d_in[1];
  const float* init_hd = (const float*)d_in[2];
  const float* W_state = (const float*)d_in[3];
  const float* b_state = (const float*)d_in[4];
  const float* W_cell  = (const float*)d_in[5];
  const float* b_cell  = (const float*)d_in[6];
  const float* W_ih    = (const float*)d_in[7];
  const float* W_hh    = (const float*)d_in[8];
  const float* b_ih    = (const float*)d_in[9];
  const float* b_hh    = (const float*)d_in[10];
  const float* W_bn    = (const float*)d_in[11];
  const float* W_pc    = (const float*)d_in[12];
  const float* b_pc    = (const float*)d_in[13];
  const float* W_hd    = (const float*)d_in[14];
  const float* b_hd    = (const float*)d_in[15];

  // workspace layout (~5 MB)
  char* ws = (char*)d_ws;
  float*    h0   = (float*)ws;    ws += (size_t)BATCH*NH*sizeof(float);
  float*    c0   = (float*)ws;    ws += (size_t)BATCH*NH*sizeof(float);
  _Float16* Wga  = (_Float16*)ws; ws += (size_t)4*NH*KA*sizeof(_Float16);
  _Float16* Wbnh = (_Float16*)ws; ws += (size_t)NBd*NH*sizeof(_Float16);
  _Float16* Wpch = (_Float16*)ws; ws += (size_t)NPCd*NBd*sizeof(_Float16);
  _Float16* Whdh = (_Float16*)ws; ws += (size_t)16*NBd*sizeof(_Float16);

  // output layout: (hd, pc, bn, hs, cs) concatenated flat
  float* o = (float*)d_out;
  float* o_hd = o;  o += (size_t)TSTEPS*BATCH*NHDd;
  float* o_pc = o;  o += (size_t)TSTEPS*BATCH*NPCd;
  float* o_bn = o;  o += (size_t)TSTEPS*BATCH*NBd;
  float* o_h  = o;  o += (size_t)TSTEPS*BATCH*NH;
  float* o_c  = o;

  prep_gate <<<(4*NH*KA + 255)/256, 256, 0, stream>>>(W_hh, W_ih, b_ih, b_hh, Wga);
  cvt_f16   <<<(NBd*NH  + 255)/256, 256, 0, stream>>>(W_bn, Wbnh, NBd*NH);
  cvt_f16   <<<(NPCd*NBd+ 255)/256, 256, 0, stream>>>(W_pc, Wpch, NPCd*NBd);
  cvt_hd_pad<<<(16*NBd  + 255)/256, 256, 0, stream>>>(W_hd, Whdh);
  init_hc   <<<(BATCH*NH)/256,      256, 0, stream>>>(init_pc, init_hd, W_state, b_state,
                                                      W_cell, b_cell, h0, c0);
  grid_lstm_main<<<BATCH/BT, NTHR, 0, stream>>>(x, b_pc, b_hd,
                                                Wga, Wbnh, Wpch, Whdh, h0, c0,
                                                o_hd, o_pc, o_bn, o_h, o_c);
}
